// StandardRNN_68066641707154
// MI455X (gfx1250) — matrix operand order repaired and checked by reading, not compiled, not run on hardware
//
#include <hip/hip_runtime.h>
#include <hip/hip_bf16.h>

// GRU: T=1024, B=64, D=256, H=512, O=256
#define T_STEPS 1024
#define BATCH   64
#define DIM     256
#define HID     512
#define GATE3   1536
#define OUTD    256

#define NCOLWG  8     // hidden-column groups (64 cols each)
#define COLS_WG 64    // hidden columns per WG
#define WG_THREADS 128 // 4 waves, one 16-col tile per wave

typedef __bf16 v16bf __attribute__((ext_vector_type(16)));
typedef __bf16 v8bf  __attribute__((ext_vector_type(8)));
typedef float  v8f   __attribute__((ext_vector_type(8)));

static __device__ __forceinline__ v8f wmma_bf16(v16bf a, v16bf b, v8f c) {
  // D = A(16x32 bf16) * B(32x16 bf16) + C(16x16 f32)
  return __builtin_amdgcn_wmma_f32_16x16x32_bf16(
      false, a, false, b, (short)0, c, false, false);
}

// A fragment: row-major A[16 x K], lda in elems, pointer pre-offset by k0.
// A 16x32 layout (matches load_b_frag): lanes 0-15 row=lane, K=0..15;
// lanes 16-31 row=lane-16, K=16..31.
static __device__ __forceinline__ v16bf load_a_frag(const __bf16* A, int lda, int lane) {
  const int row = lane & 15;
  const int kb  = (lane >> 4) * 16;
  const __bf16* p = A + (size_t)row * lda + kb;
  v8bf lo = *(const v8bf*)(p);
  v8bf hi = *(const v8bf*)(p + 8);
  v16bf r;
#pragma unroll
  for (int i = 0; i < 8; ++i) { r[i] = lo[i]; r[i + 8] = hi[i]; }
  return r;
}

// B fragment from pre-transposed Bt[N x K] row-major (ldb in elems), pre-offset
// to (n0, k0). Lanes 0-15: col N=lane, K=0..15; lanes 16-31: col N=lane-16,
// K=16..31 -> one contiguous 32B read per lane (works for global or LDS ptrs).
static __device__ __forceinline__ v16bf load_b_frag(const __bf16* Bt, int ldb, int lane) {
  const int n  = lane & 15;
  const int kb = (lane >> 4) * 16;
  return *(const v16bf*)(Bt + (size_t)n * ldb + kb);
}

// ---------------- small utility kernels ----------------

__global__ void k_cvt_bf16(const float* __restrict__ in, __bf16* __restrict__ out,
                           size_t n) {
  size_t i = (size_t)blockIdx.x * blockDim.x + threadIdx.x;
  size_t stride = (size_t)gridDim.x * blockDim.x;
  for (; i < n; i += stride) out[i] = (__bf16)in[i];
}

// in[R x C] f32 -> out[C x R] bf16
__global__ void k_transpose_cvt(const float* __restrict__ in, __bf16* __restrict__ out,
                                int R, int C) {
  size_t n = (size_t)R * C;
  size_t i = (size_t)blockIdx.x * blockDim.x + threadIdx.x;
  size_t stride = (size_t)gridDim.x * blockDim.x;
  for (; i < n; i += stride) {
    int r = (int)(i / C), c = (int)(i % C);
    out[(size_t)c * R + r] = (__bf16)in[i];
  }
}

__global__ void k_zero_u32(unsigned* p, int n) {
  int i = blockIdx.x * blockDim.x + threadIdx.x;
  if (i < n) p[i] = 0u;
}

// ---------------- recurrence ----------------
// Grid: 32 WGs = 4 batch groups (16 rows = WMMA M) x 8 column groups (64 cols).
// Each WG stages its weight slice (WiT 96KB + WhT 192KB = 288KB) into the
// 320KB WGP LDS ONCE, then runs the full T-loop with zero weight re-streaming.
// h state is the hs buffer itself: hs row t = h_{t-1} (row 0 = h0), step t
// writes row t+1. The 8 column WGs of one batch group sync per step via a
// monotonic device-scope counter (each WG +1 per step; wait for 8*(t+1)).
__global__ __launch_bounds__(WG_THREADS) void k_gru_recurrence(
    const __bf16* __restrict__ xbf,  // [T, B, D]
    const __bf16* __restrict__ WiT,  // [3H, D]
    const __bf16* __restrict__ WhT,  // [3H, H]
    const float*  __restrict__ bi,   // [3H]
    const float*  __restrict__ bhn,  // [H]
    __bf16* __restrict__ hs,         // [T+1, B, H]; row 0 pre-filled with h0
    unsigned* __restrict__ sync)     // [4] zeroed counters
{
  extern __shared__ __align__(16) __bf16 smem[];
  __bf16* wi_s = smem;                        // [3][64][DIM]  = 49152 elems
  __bf16* wh_s = smem + 3 * COLS_WG * DIM;    // [3][64][HID]  = 98304 elems

  const int tid  = threadIdx.x;
  const int lane = tid & 31;
  const int wave = tid >> 5;
  const int bg   = blockIdx.x >> 3;           // batch group 0..3
  const int cg   = blockIdx.x & 7;            // column group 0..7
  const int b0   = bg * 16;
  const int C0   = cg * COLS_WG;              // global hidden col base

  // ---- stage weight slices into LDS (once) ----
  // WiT rows g*HID + C0 + lc (g=0..2, lc=0..63), ld = DIM
  for (int e = tid * 8; e < 3 * COLS_WG * DIM; e += WG_THREADS * 8) {
    int k = e % DIM, row = e / DIM;           // row = g*64 + lc
    int g = row / COLS_WG, lc = row % COLS_WG;
    *(v8bf*)(wi_s + e) =
        *(const v8bf*)(WiT + ((size_t)(g * HID + C0 + lc)) * DIM + k);
  }
  // WhT rows g*HID + C0 + lc, ld = HID
  for (int e = tid * 8; e < 3 * COLS_WG * HID; e += WG_THREADS * 8) {
    int k = e % HID, row = e / HID;
    int g = row / COLS_WG, lc = row % COLS_WG;
    *(v8bf*)(wh_s + e) =
        *(const v8bf*)(WhT + ((size_t)(g * HID + C0 + lc)) * HID + k);
  }
  __syncthreads();

  const int lc0  = wave * 16;                 // this wave's tile inside slice
  const int ncol = lane & 15;
  const int hi4  = lane >> 4;
  const int col  = C0 + lc0 + ncol;           // absolute hidden column
  const float b_r = bi[col];
  const float b_z = bi[HID + col];
  const float b_n = bi[2 * HID + col];
  const float bh  = bhn[col];

  for (int t = 0; t < T_STEPS; ++t) {
    const __bf16* xrow = xbf + ((size_t)t * BATCH + b0) * DIM;
    const __bf16* hrow = hs + ((size_t)t * BATCH + b0) * HID;       // h_{t-1}
    __bf16*       hnew = hs + ((size_t)(t + 1) * BATCH + b0) * HID; // h_t

    if (t + 1 < T_STEPS) {
      const __bf16* xn_ = xbf + ((size_t)(t + 1) * BATCH + b0) * DIM;
      __builtin_prefetch(xn_ + (size_t)(tid & 15) * DIM, 0, 1);
    }

    v8f ar = {}, az = {}, axn = {}, ahn = {};

    // x_t @ Wi slice (fused input projection), B from LDS
#pragma unroll
    for (int k0 = 0; k0 < DIM; k0 += 32) {
      v16bf a = load_a_frag(xrow + k0, DIM, lane);
      ar  = wmma_bf16(a, load_b_frag(wi_s + (0 * COLS_WG + lc0) * DIM + k0, DIM, lane), ar);
      az  = wmma_bf16(a, load_b_frag(wi_s + (1 * COLS_WG + lc0) * DIM + k0, DIM, lane), az);
      axn = wmma_bf16(a, load_b_frag(wi_s + (2 * COLS_WG + lc0) * DIM + k0, DIM, lane), axn);
    }
    // h_{t-1} @ Wh slice, B from LDS
#pragma unroll
    for (int k0 = 0; k0 < HID; k0 += 32) {
      v16bf a = load_a_frag(hrow + k0, HID, lane);
      ar  = wmma_bf16(a, load_b_frag(wh_s + (0 * COLS_WG + lc0) * HID + k0, HID, lane), ar);
      az  = wmma_bf16(a, load_b_frag(wh_s + (1 * COLS_WG + lc0) * HID + k0, HID, lane), az);
      ahn = wmma_bf16(a, load_b_frag(wh_s + (2 * COLS_WG + lc0) * HID + k0, HID, lane), ahn);
    }

    // gates + state update; C/D layout: elem k of lane l -> (m=k+8*(l>>4), n=l&15)
#pragma unroll
    for (int k = 0; k < 8; ++k) {
      const int m = k + 8 * hi4;
      float r  = 1.0f / (1.0f + __expf(-(ar[k] + b_r)));
      float z  = 1.0f / (1.0f + __expf(-(az[k] + b_z)));
      float nn = tanhf(axn[k] + b_n + r * (ahn[k] + bh));
      float hp = (float)hrow[(size_t)m * HID + col];
      float hv = (1.0f - z) * nn + z * hp;
      hnew[(size_t)m * HID + col] = (__bf16)hv;
    }

    // per-step barrier across the 8 column groups of this batch group
    __threadfence();            // release h_t stores to device scope
    __syncthreads();
    if (tid == 0) {
      atomicAdd(&sync[bg], 1u);
      const unsigned target = (unsigned)(NCOLWG * (t + 1));
      volatile unsigned* vs = sync + bg;
      while (*vs < target) { __builtin_amdgcn_s_sleep(2); }
    }
    __syncthreads();
    __threadfence();            // acquire peers' h_t stores
  }
}

// ---------------- readout GEMM: out[T*B, O] = hs @ Wo + bo ----------------
__global__ __launch_bounds__(256) void k_gemm_out(
    const __bf16* __restrict__ hs,   // [T*B, H] (rows 1..T of state buffer)
    const __bf16* __restrict__ WoT,  // [O, H]
    const float*  __restrict__ bo,   // [O]
    float* __restrict__ out)         // [T*B, O]
{
  const int tid  = threadIdx.x;
  const int lane = tid & 31;
  const int wave = tid >> 5;
  const int m0   = blockIdx.x * 16;
  const __bf16* Arow = hs + (size_t)m0 * HID;

#pragma unroll
  for (int j = 0; j < 2; ++j) {
    const int n0 = (wave * 2 + j) * 16;
    v8f c = {};
#pragma unroll
    for (int k0 = 0; k0 < HID; k0 += 32) {
      v16bf a = load_a_frag(Arow + k0, HID, lane);
      v16bf b = load_b_frag(WoT + (size_t)n0 * HID + k0, HID, lane);
      c = wmma_bf16(a, b, c);
    }
    const int n   = lane & 15;
    const int hi4 = lane >> 4;
    const float bb = bo[n0 + n];
#pragma unroll
    for (int k = 0; k < 8; ++k) {
      const int m = m0 + k + 8 * hi4;
      out[(size_t)m * OUTD + n0 + n] = c[k] + bb;
    }
  }
}

// ---------------- host side ----------------

extern "C" void kernel_launch(void* const* d_in, const int* in_sizes, int n_in,
                              void* d_out, int out_size, void* d_ws, size_t ws_size,
                              hipStream_t stream) {
  const float* x   = (const float*)d_in[0];  // [T,B,D]
  const float* Wi  = (const float*)d_in[1];  // [D,3H]
  const float* bi  = (const float*)d_in[2];  // [3H]
  const float* Wh  = (const float*)d_in[3];  // [H,3H]
  const float* bhn = (const float*)d_in[4];  // [H]
  const float* Wo  = (const float*)d_in[5];  // [H,O]
  const float* bo  = (const float*)d_in[6];  // [O]
  const float* h0  = (const float*)d_in[7];  // [B,H]
  float* out = (float*)d_out;

  char* ws = (char*)d_ws;
  size_t off = 0;
  auto take = [&](size_t bytes) {
    char* p = ws + off;
    off += (bytes + 255) & ~(size_t)255;
    return p;
  };
  __bf16* xbf  = (__bf16*)take((size_t)T_STEPS * BATCH * DIM * 2);        // 32 MB
  __bf16* WiT  = (__bf16*)take((size_t)GATE3 * DIM * 2);                  // 768 KB
  __bf16* WhT  = (__bf16*)take((size_t)GATE3 * HID * 2);                  // 1.5 MB
  __bf16* WoT  = (__bf16*)take((size_t)OUTD * HID * 2);                   // 256 KB
  __bf16* hsb  = (__bf16*)take((size_t)(T_STEPS + 1) * BATCH * HID * 2);  // 64 MB
  unsigned* sync = (unsigned*)take(16 * sizeof(unsigned));
  (void)ws_size; (void)in_sizes; (void)n_in; (void)out_size;

  // 0) zero the step-barrier counters (every launch -> deterministic)
  k_zero_u32<<<1, 16, 0, stream>>>(sync, 16);
  // 1) convert x and h0 to bf16 (h0 -> row 0 of state buffer)
  k_cvt_bf16<<<4096, 256, 0, stream>>>(x, xbf, (size_t)T_STEPS * BATCH * DIM);
  k_cvt_bf16<<<128, 256, 0, stream>>>(h0, hsb, (size_t)BATCH * HID);
  // 2) transpose+convert weights
  k_transpose_cvt<<<1536, 256, 0, stream>>>(Wi, WiT, DIM, GATE3);
  k_transpose_cvt<<<3072, 256, 0, stream>>>(Wh, WhT, HID, GATE3);
  k_transpose_cvt<<<512, 256, 0, stream>>>(Wo, WoT, HID, OUTD);
  // 3) recurrence: 4 batch groups x 8 column groups, 288KB LDS-resident weights
  const size_t lds_bytes = (size_t)(3 * COLS_WG * DIM + 3 * COLS_WG * HID) * 2; // 294912
  k_gru_recurrence<<<4 * NCOLWG, WG_THREADS, lds_bytes, stream>>>(
      xbf, WiT, WhT, bi, bhn, hsb, sync);
  // 4) readout over h_1..h_T
  k_gemm_out<<<4096, 256, 0, stream>>>(hsb + (size_t)BATCH * HID, WoT, bo, out);
}
